// Attention_55448027791421
// MI455X (gfx1250) — matrix operand order repaired and checked by reading, not compiled, not run on hardware
//
#include <hip/hip_runtime.h>
#include <hip/hip_bf16.h>

// ---------------------------------------------------------------------------
// MI455X (gfx1250) attention pipeline, bf16 WMMA with fp32 accumulation.
//   stage 1: cvt fp32->bf16 (x, qkv_weight, proj_weight)
//   stage 2: QKV GEMM (128x128 tiles, WMMA); A tile staged by TDM
//            (tensor_load_to_lds, TENSORcnt) with HW OOB-zeroing + LDS padding
//   stage 3: flash attention per (b,h): S=QK^T (WMMA), +bias, online softmax,
//            O += P·V (WMMA); Q via TDM, V via GLOBAL_LOAD_ASYNC_TO_LDS
//   stage 4: proj GEMM (WMMA) + bias -> fp32 d_out
// ---------------------------------------------------------------------------

typedef __attribute__((ext_vector_type(16))) __bf16   bf16x16;
typedef __attribute__((ext_vector_type(8)))  float    f32x8;
typedef __attribute__((ext_vector_type(4)))  unsigned u32x4;
typedef __attribute__((ext_vector_type(8)))  unsigned u32x8;

#define BB    16
#define NN    577
#define CC    768
#define HH    12
#define HD    64
#define NPAD  640              // 577 padded to multiple of 64 (10 q-tiles of 64)
#define MROWS (BB * NN)        // 9232
#define QK_SCALE 0.125f        // 64^-0.5
#define NEG_BIG (-1.0e30f)

__device__ __forceinline__ f32x8 zero8() {
    f32x8 z;
    #pragma unroll
    for (int i = 0; i < 8; ++i) z[i] = 0.0f;
    return z;
}

// ---------------- async (ASYNCcnt) global -> LDS, 16B per lane --------------
__device__ __forceinline__ void async_load_b128(void* lds_ptr, const void* gptr) {
    const unsigned loff = (unsigned)(uintptr_t)lds_ptr;   // low 32b of generic LDS ptr = LDS offset
    asm volatile("global_load_async_to_lds_b128 %0, %1, off"
                 :: "v"(loff), "v"(gptr) : "memory");
}
__device__ __forceinline__ void wait_async() {
    asm volatile("s_wait_asynccnt 0" ::: "memory");
}

// ---------------- Tensor Data Mover (TENSORcnt) -----------------------------
// D# group 0 (128b): count=1, lds_addr, global_addr[56:0], type=2 ("image")
__device__ __forceinline__ u32x4 tdm_group0(unsigned lds_addr, unsigned long long gaddr) {
    u32x4 g;
    g[0] = 1u;                                              // count=1, user mode
    g[1] = lds_addr;                                        // [63:32]  lds_addr
    g[2] = (unsigned)(gaddr & 0xFFFFFFFFull);               // [95:64]  global_addr lo
    g[3] = (unsigned)((gaddr >> 32) & 0x01FFFFFFull)        // [120:96] global_addr hi
         | (2u << 30);                                      // [127:126] type=2
    return g;
}
// D# group 1 (256b): data_size=1 (2B), LDS padding, 2-D tile/tensor dims.
//   pad_interval: 0=2,1=4,2=8,3=16,4=32... DWORDs ; pad_amount: DWORDs-1
__device__ __forceinline__ u32x8 tdm_group1(unsigned pad_interval, unsigned pad_amount,
                                            unsigned long long tdim0, unsigned long long tdim1,
                                            unsigned tile0, unsigned tile1,
                                            unsigned long long stride0) {
    u32x8 g;
    g[0] = (1u << 16) | (1u << 20) | (pad_interval << 22) | (pad_amount << 25);
    g[1] = (unsigned)((tdim0 & 0xFFFFull) << 16);                        // tensor_dim0[15:0] @ bit48
    g[2] = (unsigned)((tdim0 >> 16) & 0xFFFFull)                         // tensor_dim0[31:16]
         | (unsigned)((tdim1 & 0xFFFFull) << 16);                        // tensor_dim1[15:0] @ bit80
    g[3] = (unsigned)((tdim1 >> 16) & 0xFFFFull)                         // tensor_dim1[31:16]
         | (tile0 << 16);                                                // tile_dim0 @ bit112
    g[4] = (tile1 & 0xFFFFu);                                            // tile_dim1; tile_dim2=0
    g[5] = (unsigned)(stride0 & 0xFFFFFFFFull);                          // tensor_dim0_stride[31:0]
    g[6] = (unsigned)((stride0 >> 32) & 0xFFFFull);                      // stride[47:32]; dim1_stride=0
    g[7] = 0u;
    return g;
}
__device__ __forceinline__ void tdm_load(u32x4 g0, u32x8 g1) {
    asm volatile("tensor_load_to_lds %0, %1" :: "s"(g0), "s"(g1) : "memory");
}
__device__ __forceinline__ void wait_tensor() {
    __builtin_amdgcn_s_wait_tensorcnt(0);
}

// ---------------- WMMA fragment helpers -------------------------------------
// A fragment 16x32 bf16 (MxK) from LDS tile laid out [M][lda]:
//   lanes 0-15: M=lane,  K in {0..7, 16..23}; lanes 16-31: M=lane-16, K in {8..15, 24..31}
__device__ __forceinline__ bf16x16 ldsA_frag(const __bf16* base, int lda, int lane) {
    const int m  = lane & 15;
    const int kb = (lane >> 4) << 3;
    const __bf16* p = base + m * lda + kb;
    bf16x16 a;
    #pragma unroll
    for (int i = 0; i < 8; ++i) { a[i] = p[i]; a[i + 8] = p[i + 16]; }
    return a;
}
// B fragment 32x16 bf16 (KxN) from LDS tile laid out [K][ldb]:
//   lanes 0-15: N=lane, K in {0..7, 16..23}; lanes 16-31: N=lane-16, K in {8..15, 24..31}
__device__ __forceinline__ bf16x16 ldsB_frag(const __bf16* base, int ldb, int lane) {
    const int n  = lane & 15;
    const int kb = (lane >> 4) << 3;
    const __bf16* p = base + kb * ldb + n;
    bf16x16 b;
    #pragma unroll
    for (int i = 0; i < 8; ++i) { b[i] = p[i * ldb]; b[i + 8] = p[(i + 16) * ldb]; }
    return b;
}
__device__ __forceinline__ f32x8 wmma_bf16(bf16x16 a, bf16x16 b, f32x8 c) {
    return __builtin_amdgcn_wmma_f32_16x16x32_bf16(false, a, false, b,
                                                   (short)0, c, false, false);
}

// ---------------------------------------------------------------------------
__global__ void cvt_f32_bf16(const float* __restrict__ s, __bf16* __restrict__ d, int n) {
    int i  = blockIdx.x * blockDim.x + threadIdx.x;
    int st = gridDim.x * blockDim.x;
    for (; i < n; i += st) d[i] = (__bf16)s[i];
}
__global__ void fill_zero_bf16(__bf16* __restrict__ d, int n) {
    int i  = blockIdx.x * blockDim.x + threadIdx.x;
    int st = gridDim.x * blockDim.x;
    for (; i < n; i += st) d[i] = (__bf16)0.0f;
}

// ---------------------------------------------------------------------------
// Tiled GEMM: C[M, Nout] = A[M, K] * Bw[Nout, K]^T  (+ epilogue)
// block = 128 threads (4 waves), 128x128 tile, wave = 64x64 (4x4 frags).
// A tile (128x32) staged by one TDM descriptor per K-step (wave 0 issues);
// hardware pads LDS rows 32->40 bf16 and zero-clamps rows past MROWS.
// ---------------------------------------------------------------------------
template <int MODE>
__global__ __launch_bounds__(128) void gemm128(
    const __bf16* __restrict__ A, const __bf16* __restrict__ Bw, int K,
    const float* __restrict__ qbias, const float* __restrict__ vbias,
    __bf16* __restrict__ Qo, __bf16* __restrict__ Ko, __bf16* __restrict__ Vo,
    const float* __restrict__ pbias, float* __restrict__ out)
{
    __shared__ __align__(16) __bf16 As[128 * 40];   // [128 rows][32 K] ld=40 (TDM-padded)
    __shared__            __bf16 Bs[32 * 136];      // [32 K][128 N] ld=136 (transposed W)

    const int tid  = threadIdx.x;
    const int lane = tid & 31;
    const int wave = tid >> 5;
    const int wy = (wave >> 1) * 64;
    const int wx = (wave & 1) * 64;
    const int bm = blockIdx.x * 128;
    const int bn = blockIdx.y * 128;
    const int rows_valid = (MROWS - bm < 128) ? (MROWS - bm) : 128;

    // pre-zero tail rows once (in case TDM drops OOB writes instead of zeroing)
    if (rows_valid < 128) {
        for (int v = tid; v < 512; v += 128) {
            const int r = v >> 2, c8 = (v & 3) << 3;
            if (r >= rows_valid)
                *reinterpret_cast<uint4*>(&As[r * 40 + c8]) = make_uint4(0u, 0u, 0u, 0u);
        }
        __syncthreads();
    }

    f32x8 acc[4][4];
    #pragma unroll
    for (int i = 0; i < 4; ++i)
        #pragma unroll
        for (int j = 0; j < 4; ++j) acc[i][j] = zero8();

    for (int k0 = 0; k0 < K; k0 += 32) {
        // ---- A tile via Tensor Data Mover: 128x32 bf16, row stride K ----
        if (wave == 0) {
            u32x4 g0 = tdm_group0((unsigned)(uintptr_t)(void*)As,
                                  (unsigned long long)(uintptr_t)(A + (size_t)bm * K + k0));
            u32x8 g1 = tdm_group1(/*pad 16DW +4DW*/ 3u, 3u,
                                  /*tensor dims*/ 32ull, (unsigned long long)rows_valid,
                                  /*tile dims*/ 32u, 128u,
                                  /*row stride*/ (unsigned long long)K);
            tdm_load(g0, g1);
        }
        // ---- B tile transposed: W[bn+n][k] -> Bs[k][n] (through VGPRs) ----
        for (int v = tid; v < 512; v += 128) {
            const int n  = v >> 2;
            const int c8 = (v & 3) << 3;
            union { uint4 u; __bf16 h[8]; } t;
            t.u = *reinterpret_cast<const uint4*>(Bw + (size_t)(bn + n) * K + k0 + c8);
            #pragma unroll
            for (int i = 0; i < 8; ++i) Bs[(c8 + i) * 136 + n] = t.h[i];
            if (k0 + 32 < K)
                __builtin_prefetch(Bw + (size_t)(bn + n) * K + k0 + 32 + c8, 0, 0);
        }
        wait_tensor();
        __syncthreads();

        bf16x16 af[4], bfr[4];
        #pragma unroll
        for (int fi = 0; fi < 4; ++fi)
            af[fi] = ldsA_frag(As + (wy + fi * 16) * 40, 40, lane);
        #pragma unroll
        for (int fj = 0; fj < 4; ++fj)
            bfr[fj] = ldsB_frag(Bs + wx + fj * 16, 136, lane);
        #pragma unroll
        for (int fi = 0; fi < 4; ++fi)
            #pragma unroll
            for (int fj = 0; fj < 4; ++fj)
                acc[fi][fj] = wmma_bf16(af[fi], bfr[fj], acc[fi][fj]);
        __syncthreads();
    }

    // ---- epilogue: C/D layout lane holds col N=lane&15, rows r+8*(lane>>4) ----
    const int cl = lane & 15;
    const int rh = (lane >> 4) * 8;
    #pragma unroll
    for (int fi = 0; fi < 4; ++fi) {
        #pragma unroll
        for (int r = 0; r < 8; ++r) {
            const int grow = bm + wy + fi * 16 + rh + r;
            if (grow >= MROWS) continue;
            if constexpr (MODE == 0) {
                const int b_ = grow / NN;
                const int n_ = grow - b_ * NN;
                #pragma unroll
                for (int fj = 0; fj < 4; ++fj) {
                    const int gcol = bn + wx + fj * 16 + cl;     // [0, 2304)
                    const int s_   = gcol / CC;                  // 0=Q 1=K 2=V
                    const int rem  = gcol - s_ * CC;
                    const int h_   = rem >> 6;
                    const int d_   = rem & 63;
                    float v = acc[fi][fj][r];
                    if (s_ == 0)      v = (v + qbias[rem]) * QK_SCALE;
                    else if (s_ == 2) v = v + vbias[rem];
                    __bf16* dst = (s_ == 0) ? Qo : ((s_ == 1) ? Ko : Vo);
                    dst[(((size_t)b_ * HH + h_) * NPAD + n_) * HD + d_] = (__bf16)v;
                }
            } else {
                #pragma unroll
                for (int fj = 0; fj < 4; ++fj) {
                    const int gcol = bn + wx + fj * 16 + cl;
                    out[(size_t)grow * CC + gcol] = acc[fi][fj][r] + pbias[gcol];
                }
            }
        }
    }
}

// ---------------------------------------------------------------------------
// Flash attention: grid (B*H, NPAD/64), block 128 (4 waves, 16 q-rows each).
// Q tile via TDM; V tiles via async global->LDS; K staged transposed.
// ---------------------------------------------------------------------------
__global__ __launch_bounds__(128) void attn_flash(
    const __bf16* __restrict__ Qb, const __bf16* __restrict__ Kb,
    const __bf16* __restrict__ Vb, const float* __restrict__ bias,
    __bf16* __restrict__ ctx)
{
    __shared__ __align__(16) __bf16 Qs[64 * 80];   // [q-row][HD]   ld=80 (TDM-padded)
    __shared__            __bf16 Ks[64 * 40];      // [HD][key]     ld=40 (transposed)
    __shared__ __align__(16) __bf16 Vs[32 * 80];   // [key][HD]     ld=80
    __shared__            __bf16 Ps[64 * 40];      // [q-row][key]  ld=40

    const int tid  = threadIdx.x;
    const int lane = tid & 31;
    const int wave = tid >> 5;
    const int bh = blockIdx.x;               // 0..191
    const int b_ = bh / HH;
    const int h_ = bh - b_ * HH;
    const int q0 = blockIdx.y * 64;
    const size_t base = (size_t)bh * NPAD * HD;

    // stage Q tile [64][64] via TDM (rows always within zero-filled NPAD buffer)
    if (wave == 0) {
        u32x4 g0 = tdm_group0((unsigned)(uintptr_t)(void*)Qs,
                              (unsigned long long)(uintptr_t)(Qb + base + (size_t)q0 * HD));
        u32x8 g1 = tdm_group1(/*pad 32DW +8DW*/ 4u, 7u,
                              64ull, 64ull, 64u, 64u, (unsigned long long)HD);
        tdm_load(g0, g1);
    }

    f32x8 o[4];
    float m_i[8], l_i[8];
    #pragma unroll
    for (int f = 0; f < 4; ++f) o[f] = zero8();
    #pragma unroll
    for (int r = 0; r < 8; ++r) { m_i[r] = NEG_BIG; l_i[r] = 0.0f; }

    const int cl = lane & 15;
    const int rh = (lane >> 4) * 8;
    const int qrow0 = q0 + wave * 16 + rh;

    for (int j0 = 0; j0 < NN; j0 += 32) {
        __syncthreads();   // previous PV done before K,V overwrite
        // K tile transposed into Ks[hd][key] (through VGPRs)
        for (int v = tid; v < 256; v += 128) {
            const int kr = v >> 3;
            const int c8 = (v & 7) << 3;
            union { uint4 u; __bf16 h[8]; } t;
            t.u = *reinterpret_cast<const uint4*>(Kb + base + (size_t)(j0 + kr) * HD + c8);
            #pragma unroll
            for (int i = 0; i < 8; ++i) Ks[(c8 + i) * 40 + kr] = t.h[i];
        }
        // V tile straight copy Vs[key][hd] via async DMA
        for (int v = tid; v < 256; v += 128) {
            const int kr = v >> 3;
            const int c8 = (v & 7) << 3;
            async_load_b128(&Vs[kr * 80 + c8], Vb + base + (size_t)(j0 + kr) * HD + c8);
        }
        wait_async();
        wait_tensor();    // Q tile (first iteration only; no-op afterwards)
        __syncthreads();

        // S = Q K^T : 16 rows x 32 keys per wave, K-dim = HD = 2 steps of 32
        f32x8 s[2];
        s[0] = zero8(); s[1] = zero8();
        #pragma unroll
        for (int ks = 0; ks < 2; ++ks) {
            bf16x16 a  = ldsA_frag(Qs + (wave * 16) * 80 + ks * 32, 80, lane);
            bf16x16 b0 = ldsB_frag(Ks + (ks * 32) * 40 +  0, 40, lane);
            bf16x16 b1 = ldsB_frag(Ks + (ks * 32) * 40 + 16, 40, lane);
            s[0] = wmma_bf16(a, b0, s[0]);
            s[1] = wmma_bf16(a, b1, s[1]);
        }

        // + relative position bias, mask padded key columns
        #pragma unroll
        for (int nf = 0; nf < 2; ++nf) {
            const int col = j0 + nf * 16 + cl;
            #pragma unroll
            for (int r = 0; r < 8; ++r) {
                const int row = qrow0 + r;
                float v = s[nf][r];
                if (col < NN) {
                    if (row < NN) v += bias[((size_t)h_ * NN + row) * NN + col];
                } else {
                    v = NEG_BIG;
                }
                s[nf][r] = v;
            }
        }

        // online softmax (row spread over 16-lane half + 2 fragments)
        #pragma unroll
        for (int r = 0; r < 8; ++r) {
            float mx = fmaxf(s[0][r], s[1][r]);
            #pragma unroll
            for (int off = 8; off > 0; off >>= 1) mx = fmaxf(mx, __shfl_xor(mx, off, 16));
            const float mnew  = fmaxf(m_i[r], mx);
            const float scale = __expf(m_i[r] - mnew);
            const float p0 = __expf(s[0][r] - mnew);
            const float p1 = __expf(s[1][r] - mnew);
            s[0][r] = p0; s[1][r] = p1;
            float rs = p0 + p1;
            #pragma unroll
            for (int off = 8; off > 0; off >>= 1) rs += __shfl_xor(rs, off, 16);
            l_i[r] = l_i[r] * scale + rs;
            m_i[r] = mnew;
            #pragma unroll
            for (int f = 0; f < 4; ++f) o[f][r] *= scale;
        }

        // P -> LDS (D-layout to A-layout repack via memory)
        #pragma unroll
        for (int nf = 0; nf < 2; ++nf) {
            const int col = nf * 16 + cl;
            #pragma unroll
            for (int r = 0; r < 8; ++r)
                Ps[(wave * 16 + rh + r) * 40 + col] = (__bf16)s[nf][r];
        }
        __syncthreads();

        // O += P(16x32) * V(32x64)
        {
            bf16x16 a = ldsA_frag(Ps + (wave * 16) * 40, 40, lane);
            #pragma unroll
            for (int f = 0; f < 4; ++f) {
                bf16x16 b = ldsB_frag(Vs + f * 16, 80, lane);
                o[f] = wmma_bf16(a, b, o[f]);
            }
        }
    }

    // epilogue: ctx[b, n, h*64 + d] bf16
    #pragma unroll
    for (int r = 0; r < 8; ++r) {
        const int row = qrow0 + r;
        if (row >= NN) continue;
        const float inv = 1.0f / l_i[r];
        #pragma unroll
        for (int f = 0; f < 4; ++f)
            ctx[((size_t)(b_ * NN + row)) * CC + h_ * HD + f * 16 + cl] =
                (__bf16)(o[f][r] * inv);
    }
}

// ---------------------------------------------------------------------------
extern "C" void kernel_launch(void* const* d_in, const int* in_sizes, int n_in,
                              void* d_out, int out_size, void* d_ws, size_t ws_size,
                              hipStream_t stream) {
    const float* x     = (const float*)d_in[0];
    const float* rpb   = (const float*)d_in[1];
    const float* wqkv  = (const float*)d_in[2];
    const float* qbias = (const float*)d_in[3];
    const float* vbias = (const float*)d_in[4];
    const float* wproj = (const float*)d_in[5];
    const float* pbias = (const float*)d_in[6];
    float* out = (float*)d_out;

    // workspace carve (each region 256B aligned); total ~81 MB
    char* p = (char*)d_ws;
    auto carve = [&](size_t bytes) -> char* {
        char* r = p;
        p += (bytes + 255) & ~(size_t)255;
        return r;
    };
    const size_t qkvElems = (size_t)BB * HH * NPAD * HD;   // 7,864,320
    __bf16* xb  = (__bf16*)carve((size_t)MROWS * CC * 2);
    __bf16* wqb = (__bf16*)carve((size_t)3 * CC * CC * 2);
    __bf16* wpb = (__bf16*)carve((size_t)CC * CC * 2);
    __bf16* Qb  = (__bf16*)carve(qkvElems * 2);
    __bf16* Kb  = (__bf16*)carve(qkvElems * 2);
    __bf16* Vb  = (__bf16*)carve(qkvElems * 2);
    __bf16* ctx = (__bf16*)carve((size_t)MROWS * CC * 2);
    (void)ws_size; (void)in_sizes; (void)n_in; (void)out_size;

    // stage 1: fp32 -> bf16
    cvt_f32_bf16<<<2048, 256, 0, stream>>>(x,     xb,  MROWS * CC);
    cvt_f32_bf16<<<2048, 256, 0, stream>>>(wqkv,  wqb, 3 * CC * CC);
    cvt_f32_bf16<<<1024, 256, 0, stream>>>(wproj, wpb, CC * CC);
    // zero-fill padded Q/K/V (covers rows 577..639 against 0xAA poison)
    fill_zero_bf16<<<2048, 256, 0, stream>>>(Qb, (int)qkvElems);
    fill_zero_bf16<<<2048, 256, 0, stream>>>(Kb, (int)qkvElems);
    fill_zero_bf16<<<2048, 256, 0, stream>>>(Vb, (int)qkvElems);

    // stage 2: QKV GEMM [9232x768]x[768x2304]^T, 128x128 tiles
    gemm128<0><<<dim3((MROWS + 127) / 128, (3 * CC) / 128), 128, 0, stream>>>(
        xb, wqb, CC, qbias, vbias, Qb, Kb, Vb, nullptr, nullptr);

    // stage 3: flash attention, grid (B*H, NPAD/64)
    attn_flash<<<dim3(BB * HH, NPAD / 64), 128, 0, stream>>>(Qb, Kb, Vb, rpb, ctx);

    // stage 4: proj GEMM [9232x768]x[768x768]^T + bias -> fp32 out
    gemm128<1><<<dim3((MROWS + 127) / 128, CC / 128), 128, 0, stream>>>(
        ctx, wpb, CC, nullptr, nullptr, nullptr, nullptr, nullptr, pbias, out);
}
